// multiHeadAttention_6038724018928
// MI455X (gfx1250) — matrix operand order repaired and checked by reading, not compiled, not run on hardware
//
#include <hip/hip_runtime.h>

// ---------- problem constants ----------
#define B_  8
#define S_  1024
#define E_  768
#define H_  12
#define D_  64
#define NQKV 2304          // 3*H*D
#define MTOT (B_*S_)       // 8192

typedef __attribute__((ext_vector_type(16))) __bf16 v16bf;
typedef __attribute__((ext_vector_type(8)))  __bf16 v8bf;
typedef __attribute__((ext_vector_type(8)))  float  v8f;

// float -> bf16 bits (round to nearest even)
static __device__ __forceinline__ unsigned short f2bf(float f) {
    union { float f; unsigned u; } x; x.f = f;
    unsigned u = x.u + 0x7fffu + ((x.u >> 16) & 1u);
    return (unsigned short)(u >> 16);
}

// A-fragment (16x32 bf16, MxK): lane m = L%16, half = L/16.
// elems 0..15 : K = half*16 + 0..15
static __device__ __forceinline__ v16bf load_a(const unsigned short* p, int half) {
    v8bf g0 = *(const v8bf*)(p + half * 16);
    v8bf g1 = *(const v8bf*)(p + half * 16 + 8);
    return __builtin_shufflevector(g0, g1, 0,1,2,3,4,5,6,7,8,9,10,11,12,13,14,15);
}

// B-fragment (32x16 bf16, KxN): lane n = L%16, covers K = (L/16)*16 .. +15,
// 16 consecutive bf16 down column n (column stored contiguously).
static __device__ __forceinline__ v16bf load_b16(const unsigned short* p) {
    v8bf g0 = *(const v8bf*)(p);
    v8bf g1 = *(const v8bf*)(p + 8);
    return __builtin_shufflevector(g0, g1, 0,1,2,3,4,5,6,7,8,9,10,11,12,13,14,15);
}

static __device__ __forceinline__ v8f wmma_bf16(v16bf a, v16bf b, v8f c) {
    return __builtin_amdgcn_wmma_f32_16x16x32_bf16(false, a, false, b, (short)0, c, false, false);
}

// ---------- CDNA5 async copy helpers ----------
static __device__ __forceinline__ unsigned to_lds_off(const void* p) {
    return (unsigned)(unsigned long long)
        (const __attribute__((address_space(3))) void*)p;
}
// GLOBAL_LOAD_ASYNC_TO_LDS_B128: per-lane LDS[vdst] = MEM[vaddr] (ASYNCcnt)
static __device__ __forceinline__ void async_ld_b128(unsigned lds_off,
                                                     const void* gaddr) {
    asm volatile("global_load_async_to_lds_b128 %0, %1, off"
                 :: "v"(lds_off), "v"((unsigned long long)(uintptr_t)gaddr)
                 : "memory");
}
static __device__ __forceinline__ void wait_async0() {
#if __has_builtin(__builtin_amdgcn_s_wait_asynccnt)
    __builtin_amdgcn_s_wait_asynccnt(0);
#else
    asm volatile("s_wait_asynccnt 0x0" ::: "memory");
#endif
}

// ---------- prep kernels ----------
__global__ void cvt_hidden_kernel(const float* __restrict__ x,
                                  unsigned short* __restrict__ y, int n) {
    int i = blockIdx.x * blockDim.x + threadIdx.x;
    if (i < n) y[i] = f2bf(x[i]);
}

// Wqkv_t[n][e], n in [0,2304): region = n/768 (Q,K,V), h=(n%768)/64, d=n%64
__global__ void pack_wqkv_kernel(const float* __restrict__ Wq,
                                 const float* __restrict__ Wk,
                                 const float* __restrict__ Wv,
                                 unsigned short* __restrict__ t) {
    int i = blockIdx.x * blockDim.x + threadIdx.x;
    if (i >= NQKV * E_) return;
    int n = i / E_, e = i % E_;
    int region = n / E_;
    int hl = n % E_;
    int h = hl >> 6, d = hl & 63;
    const float* W = (region == 0) ? Wq : (region == 1) ? Wk : Wv;
    t[i] = f2bf(W[(h * E_ + e) * D_ + d]);
}

// Wo_t[n][k] = Wo[k][n]
__global__ void pack_wo_kernel(const float* __restrict__ Wo,
                               unsigned short* __restrict__ t) {
    int i = blockIdx.x * blockDim.x + threadIdx.x;
    if (i >= E_ * E_) return;
    int n = i / E_, k = i % E_;
    t[i] = f2bf(Wo[k * E_ + n]);
}

// =====================================================================
// Block-tiled GEMM core: block = 4 waves = 32x256 output, wave = 32x64.
// A (32 rows x 64 k) double-buffer staged through LDS with async b128
// loads; 16 WMMA per wave per stage (one barrier per 64 k).
// =====================================================================
#define GEMM_BODY(Xb, Wt, bm, bn, atile, acc)                                    \
    {                                                                            \
        int tid  = threadIdx.x;                                                  \
        int lane = tid & 31;                                                     \
        int w    = tid >> 5;                                                     \
        int half = lane >> 4, n16 = lane & 15;                                   \
        int srow = tid >> 2, skc = tid & 3; /* staging: row 0..31, chunk 0..3 */ \
        const unsigned short* xrow = (Xb) + ((size_t)((bm) * 32 + srow)) * E_;   \
        unsigned sd00 = to_lds_off(&atile[0][srow][skc * 8]);                    \
        unsigned sd01 = to_lds_off(&atile[0][srow][(skc + 4) * 8]);              \
        unsigned sd10 = to_lds_off(&atile[1][srow][skc * 8]);                    \
        unsigned sd11 = to_lds_off(&atile[1][srow][(skc + 4) * 8]);              \
        int ncol0 = (bn) * 256 + w * 64;                                         \
        async_ld_b128(sd00, xrow + skc * 8);          /* stage kb=0 -> buf0 */   \
        async_ld_b128(sd01, xrow + (skc + 4) * 8);                               \
        int buf = 0;                                                             \
        for (int kbi = 0; kbi < E_ / 64; ++kbi) {                                \
            int kb = kbi * 64;                                                   \
            wait_async0();                                                       \
            __syncthreads();                                                     \
            if (kbi + 1 < E_ / 64) {                                             \
                async_ld_b128(buf ? sd00 : sd10, xrow + kb + 64 + skc * 8);      \
                async_ld_b128(buf ? sd01 : sd11, xrow + kb + 64 + (skc + 4) * 8);\
            }                                                                    \
            _Pragma("unroll")                                                    \
            for (int kk = 0; kk < 64; kk += 32) {                                \
                v16bf af0 = load_a(&atile[buf][n16][kk], half);                  \
                v16bf af1 = load_a(&atile[buf][16 + n16][kk], half);             \
                _Pragma("unroll")                                                \
                for (int j = 0; j < 4; ++j) {                                    \
                    const unsigned short* bcol = (Wt) +                          \
                        (size_t)(ncol0 + j * 16 + n16) * E_ + kb + kk + half*16; \
                    __builtin_prefetch(bcol + 128, 0, 1);                        \
                    v16bf bf = load_b16(bcol);                                   \
                    acc[0][j] = wmma_bf16(af0, bf, acc[0][j]);                   \
                    acc[1][j] = wmma_bf16(af1, bf, acc[1][j]);                   \
                }                                                                \
            }                                                                    \
            buf ^= 1;                                                            \
        }                                                                        \
    }

// ---------- QKV projection: [8192 x 768] x [768 x 2304] ----------
__global__ void __launch_bounds__(128, 1)
qkv_gemm_kernel(const unsigned short* __restrict__ Xb,
                const unsigned short* __restrict__ Wt,
                const float* __restrict__ bq,
                const float* __restrict__ bk,
                const float* __restrict__ bv,
                unsigned short* __restrict__ Qb,
                unsigned short* __restrict__ Kb,
                unsigned short* __restrict__ Vt) {
    __shared__ __align__(16) unsigned short atile[2][32][64];
    int bm = blockIdx.x / (NQKV / 256);   // 0..255
    int bn = blockIdx.x % (NQKV / 256);   // 0..8
    int lane = threadIdx.x & 31;
    int w    = threadIdx.x >> 5;
    int half = lane >> 4, n16 = lane & 15;

    v8f acc[2][4] = {};
    GEMM_BODY(Xb, Wt, bm, bn, atile, acc);

#pragma unroll
    for (int t = 0; t < 2; ++t) {
#pragma unroll
        for (int j = 0; j < 4; ++j) {
            int n = bn * 256 + w * 64 + j * 16 + n16;
            int region = n / E_;
            int hl = n % E_;
            int h = hl >> 6, d = hl & 63;
#pragma unroll
            for (int v = 0; v < 8; ++v) {
                int m = bm * 32 + t * 16 + v + 8 * half;   // global row in B*S
                int b = m >> 10, s = m & 1023;
                float val = acc[t][j][v];
                if (region == 0) {
                    val = (val + bq[hl]) * 0.125f;         // fold 1/sqrt(64)
                    Qb[((size_t)(b * H_ + h) * S_ + s) * D_ + d] = f2bf(val);
                } else if (region == 1) {
                    val += bk[hl];
                    Kb[((size_t)(b * H_ + h) * S_ + s) * D_ + d] = f2bf(val);
                } else {
                    val += bv[hl];
                    Vt[((size_t)(b * H_ + h) * D_ + d) * S_ + s] = f2bf(val);
                }
            }
        }
    }
}

// ---------- output projection: [8192 x 768] x [768 x 768] + bo ----------
__global__ void __launch_bounds__(128, 1)
out_gemm_kernel(const unsigned short* __restrict__ Ab,
                const unsigned short* __restrict__ Wt,
                const float* __restrict__ bo,
                float* __restrict__ out) {
    __shared__ __align__(16) unsigned short atile[2][32][64];
    int bm = blockIdx.x / (E_ / 256);     // 0..255
    int bn = blockIdx.x % (E_ / 256);     // 0..2
    int lane = threadIdx.x & 31;
    int w    = threadIdx.x >> 5;
    int half = lane >> 4, n16 = lane & 15;

    v8f acc[2][4] = {};
    GEMM_BODY(Ab, Wt, bm, bn, atile, acc);

#pragma unroll
    for (int t = 0; t < 2; ++t) {
#pragma unroll
        for (int j = 0; j < 4; ++j) {
            int n = bn * 256 + w * 64 + j * 16 + n16;
            float bias = bo[n];
#pragma unroll
            for (int v = 0; v < 8; ++v) {
                int m = bm * 32 + t * 16 + v + 8 * half;
                out[(size_t)m * E_ + n] = acc[t][j][v] + bias;
            }
        }
    }
}

// ---------- flash attention: one (b,h,16-row q tile) per wave ----------
__global__ void __launch_bounds__(128, 1)
attn_kernel(const unsigned short* __restrict__ Qb,
            const unsigned short* __restrict__ Kb,
            const unsigned short* __restrict__ Vt,
            unsigned short* __restrict__ attn) {
    __shared__ __align__(16) unsigned short plds[4][16 * 40]; // P staging, stride 40
    int w    = threadIdx.x >> 5;
    int wid  = (blockIdx.x * blockDim.x + threadIdx.x) >> 5;
    int lane = threadIdx.x & 31;
    int half = lane >> 4, n16 = lane & 15;

    int b  = wid / (H_ * (S_ / 16));
    int r  = wid % (H_ * (S_ / 16));
    int h  = r / (S_ / 16);
    int qt = r % (S_ / 16);

    const size_t bh = (size_t)(b * H_ + h);
    const unsigned short* qrow = Qb + (bh * S_ + qt * 16 + n16) * D_;
    v16bf qa0 = load_a(qrow, half);       // d 0..31
    v16bf qa1 = load_a(qrow + 32, half);  // d 32..63

    v8f o0 = {}, o1 = {}, o2 = {}, o3 = {};
    float rmax[8], rsum[8];
#pragma unroll
    for (int v = 0; v < 8; ++v) { rmax[v] = -3.0e38f; rsum[v] = 0.0f; }

    for (int tb = 0; tb < S_ / 32; ++tb) {
        int tbase = tb * 32;
        // ---- scores: two 16-col tiles ----
        const unsigned short* kr0 = Kb + (bh * S_ + tbase + n16) * D_;
        const unsigned short* kr1 = kr0 + 16 * D_;
        v8f s0 = {}, s1 = {};
        s0 = wmma_bf16(qa0, load_b16(kr0 + half * 16), s0);
        s0 = wmma_bf16(qa1, load_b16(kr0 + 32 + half * 16), s0);
        s1 = wmma_bf16(qa0, load_b16(kr1 + half * 16), s1);
        s1 = wmma_bf16(qa1, load_b16(kr1 + 32 + half * 16), s1);

        // ---- online softmax (rows live on VGPR v within 16-lane half) ----
#pragma unroll
        for (int v = 0; v < 8; ++v) {
            float tmax = fmaxf(s0[v], s1[v]);
#pragma unroll
            for (int m = 1; m < 16; m <<= 1)
                tmax = fmaxf(tmax, __shfl_xor(tmax, m, 32));
            float newm  = fmaxf(rmax[v], tmax);
            float scale = __expf(rmax[v] - newm);
            rmax[v] = newm;
            float p0 = __expf(s0[v] - newm);
            float p1 = __expf(s1[v] - newm);
            float ps = p0 + p1;
#pragma unroll
            for (int m = 1; m < 16; m <<= 1)
                ps += __shfl_xor(ps, m, 32);
            rsum[v] = rsum[v] * scale + ps;
            o0[v] *= scale; o1[v] *= scale; o2[v] *= scale; o3[v] *= scale;
            int row = v + 8 * half;
            plds[w][row * 40 + n16]      = f2bf(p0);
            plds[w][row * 40 + 16 + n16] = f2bf(p1);
        }
        __syncthreads();
        // ---- P (16x32) as A-fragment from LDS ----
        const unsigned short* prow = &plds[w][n16 * 40];
        v8bf g0 = *(const v8bf*)(prow + half * 16);
        v8bf g1 = *(const v8bf*)(prow + half * 16 + 8);
        v16bf pa = __builtin_shufflevector(g0, g1,
                     0,1,2,3,4,5,6,7,8,9,10,11,12,13,14,15);
        // ---- O += P x V (V transposed: [B,H,D,S]) ----
        const unsigned short* vb = Vt + (bh * D_ + n16) * S_ + tbase + half * 16;
        o0 = wmma_bf16(pa, load_b16(vb), o0);
        o1 = wmma_bf16(pa, load_b16(vb + (size_t)16 * S_), o1);
        o2 = wmma_bf16(pa, load_b16(vb + (size_t)32 * S_), o2);
        o3 = wmma_bf16(pa, load_b16(vb + (size_t)48 * S_), o3);
        __syncthreads();
    }

    // ---- normalize & store attn [B,S,H*D] as bf16 ----
#pragma unroll
    for (int v = 0; v < 8; ++v) {
        float inv = 1.0f / rsum[v];
        int s = qt * 16 + v + 8 * half;
        size_t base = ((size_t)b * S_ + s) * E_ + h * D_ + n16;
        attn[base]      = f2bf(o0[v] * inv);
        attn[base + 16] = f2bf(o1[v] * inv);
        attn[base + 32] = f2bf(o2[v] * inv);
        attn[base + 48] = f2bf(o3[v] * inv);
    }
}

// ---------- launcher ----------
extern "C" void kernel_launch(void* const* d_in, const int* in_sizes, int n_in,
                              void* d_out, int out_size, void* d_ws, size_t ws_size,
                              hipStream_t stream) {
    const float* hidden = (const float*)d_in[0];
    const float* Wq = (const float*)d_in[1];
    const float* Wk = (const float*)d_in[2];
    const float* Wv = (const float*)d_in[3];
    const float* bq = (const float*)d_in[4];
    const float* bk = (const float*)d_in[5];
    const float* bv = (const float*)d_in[6];
    const float* Wo = (const float*)d_in[7];
    const float* bo = (const float*)d_in[8];

    char* ws = (char*)d_ws;
    const size_t SZ_X   = (size_t)MTOT * E_ * 2;        // 12,582,912
    const size_t SZ_WQ  = (size_t)NQKV * E_ * 2;        //  3,538,944
    const size_t SZ_WO  = (size_t)E_ * E_ * 2;          //  1,179,648
    const size_t SZ_QKV = (size_t)B_ * H_ * S_ * D_ * 2;// 12,582,912
    unsigned short* Xb   = (unsigned short*)(ws);
    unsigned short* Wqkv = (unsigned short*)(ws + SZ_X);
    unsigned short* Wot  = (unsigned short*)(ws + SZ_X + SZ_WQ);
    unsigned short* Qb   = (unsigned short*)(ws + SZ_X + SZ_WQ + SZ_WO);
    unsigned short* Kb   = (unsigned short*)(ws + SZ_X + SZ_WQ + SZ_WO + SZ_QKV);
    unsigned short* Vt   = (unsigned short*)(ws + SZ_X + SZ_WQ + SZ_WO + 2 * SZ_QKV);
    unsigned short* Attn = (unsigned short*)(ws + SZ_X + SZ_WQ + SZ_WO + 3 * SZ_QKV);
    float* out = (float*)d_out;

    // 1) hidden -> bf16
    {
        int n = MTOT * E_;
        cvt_hidden_kernel<<<(n + 255) / 256, 256, 0, stream>>>(hidden, Xb, n);
    }
    // 2) pack weights
    {
        int n = NQKV * E_;
        pack_wqkv_kernel<<<(n + 255) / 256, 256, 0, stream>>>(Wq, Wk, Wv, Wqkv);
        int n2 = E_ * E_;
        pack_wo_kernel<<<(n2 + 255) / 256, 256, 0, stream>>>(Wo, Wot);
    }
    // 3) QKV projection: block = 32x256 tile, 4 waves of 32x64
    {
        int blocks = (MTOT / 32) * (NQKV / 256);   // 256*9 = 2304
        qkv_gemm_kernel<<<blocks, 128, 0, stream>>>(Xb, Wqkv, bq, bk, bv, Qb, Kb, Vt);
    }
    // 4) attention: B*H*(S/16) = 6144 waves
    {
        int waves = B_ * H_ * (S_ / 16);
        attn_kernel<<<waves / 4, 128, 0, stream>>>(Qb, Kb, Vt, Attn);
    }
    // 5) output projection: block = 32x256 tile
    {
        int blocks = (MTOT / 32) * (E_ / 256);     // 256*3 = 768
        out_gemm_kernel<<<blocks, 128, 0, stream>>>(Attn, Wot, bo, out);
    }
    (void)in_sizes; (void)n_in; (void)out_size; (void)ws_size;
}
